// LSTM_33148557590883
// MI455X (gfx1250) — matrix operand order repaired and checked by reading, not compiled, not run on hardware
//
#include <hip/hip_runtime.h>
#include <hip/hip_bf16.h>

// ---------------------------------------------------------------------------
// LSTM cell, B=16384, IN=OUT=1024, K=INC=2048, gates N=4096.
// Phase 1: pack x|h -> A (bf16, WMMA A-fragment layout),
//          Wf/Wu/Wc/Wo -> B (bf16, WMMA B-fragment layout).
// Phase 2: bf16 WMMA GEMM (f32 accum), async-to-LDS double-buffered staging,
//          fused LSTM epilogue. Workspace: A 64MB + B 16MB (ws >= 80 MB).
// ---------------------------------------------------------------------------

typedef __attribute__((ext_vector_type(16))) __bf16 v16bf;
typedef __attribute__((ext_vector_type(8)))  float  v8f;

#define BATCH   16384
#define KDIM    2048
#define OUTF    1024
#define KCHUNKS 64          // KDIM / 32
#define RT_TOT  1024        // BATCH / 16 row tiles
#define CT_TOT  256         // 4096 / 16 col tiles (W-row tiles)

__device__ __forceinline__ unsigned short f2bf(float f) {
    unsigned int u = __float_as_uint(f);
    unsigned int r = u + 0x7FFFu + ((u >> 16) & 1u);   // round-to-nearest-even
    return (unsigned short)(r >> 16);
}

__device__ __forceinline__ float sigf(float x) {
    return 1.0f / (1.0f + __expf(-x));
}
__device__ __forceinline__ float tanh_fast(float x) {
    float e = __expf(-2.0f * fabsf(x));
    float t = (1.0f - e) / (1.0f + e);
    return copysignf(t, x);
}

// CDNA5 async copy: per-lane 16B global -> LDS, tracked by ASYNCcnt.
// INST_OFFSET is added to BOTH the LDS and global addresses (ISA 08 §4.4),
// so the +16 pair member needs no extra address arithmetic.
__device__ __forceinline__ void async_b128(unsigned lds_addr, const void* gaddr) {
    asm volatile("global_load_async_to_lds_b128 %0, %1, off"
                 :: "v"(lds_addr), "v"(gaddr) : "memory");
}
__device__ __forceinline__ void async_b128_o16(unsigned lds_addr, const void* gaddr) {
    asm volatile("global_load_async_to_lds_b128 %0, %1, off offset:16"
                 :: "v"(lds_addr), "v"(gaddr) : "memory");
}
__device__ __forceinline__ void wait_async0() {
    asm volatile("s_wait_asynccnt 0x0" ::: "memory");
}

// ---------------------------------------------------------------------------
// Pack A = concat(x, h) [16384 x 2048] f32 -> bf16 fragments.
// Fragment (kc, rt) = 1KB: lane l (0..31) holds 32B:
//   row = rt*16 + (l&15); half h=0: K = kc*32 + (l>>4)*8 + {0..7}; h=1: +16.
// ---------------------------------------------------------------------------
__global__ __launch_bounds__(256) void pack_A_kernel(
    const float* __restrict__ x, const float* __restrict__ h,
    unsigned short* __restrict__ Ap)
{
    unsigned tid  = blockIdx.x * 256u + threadIdx.x;   // 2^22 threads
    unsigned hh   = tid & 1u;
    unsigned lane = (tid >> 1) & 31u;
    unsigned rt   = (tid >> 6) & 1023u;
    unsigned kc   = tid >> 16;
    unsigned row  = rt * 16u + (lane & 15u);
    unsigned k0   = kc * 32u + ((lane >> 4) * 8u) + hh * 16u;

    const float* src = (k0 < 1024u) ? (x + (size_t)row * 1024u + k0)
                                    : (h + (size_t)row * 1024u + (k0 - 1024u));
    unsigned short t[8];
#pragma unroll
    for (int i = 0; i < 8; ++i) t[i] = f2bf(__builtin_nontemporal_load(src + i));

    uint4 v;
    v.x = (unsigned)t[0] | ((unsigned)t[1] << 16);
    v.y = (unsigned)t[2] | ((unsigned)t[3] << 16);
    v.z = (unsigned)t[4] | ((unsigned)t[5] << 16);
    v.w = (unsigned)t[6] | ((unsigned)t[7] << 16);
    size_t dst = (((size_t)kc * RT_TOT + rt) * 32u + lane) * 16u + hh * 8u; // ushort units
    *(uint4*)(Ap + dst) = v;
}

// ---------------------------------------------------------------------------
// Pack B = stacked [Wf;Wu;Wc;Wo] [4096 x 2048] f32 -> bf16 fragments.
// Fragment (kc, ct) = 1KB: lane l holds 32B:
//   n = ct*16 + (l&15); K = kc*32 + (l>>4)*8 + h*16 + {0..7}.
// ---------------------------------------------------------------------------
__global__ __launch_bounds__(256) void pack_B_kernel(
    const float* __restrict__ Wf, const float* __restrict__ Wu,
    const float* __restrict__ Wc, const float* __restrict__ Wo,
    unsigned short* __restrict__ Bp)
{
    unsigned tid  = blockIdx.x * 256u + threadIdx.x;   // 2^20 threads
    unsigned hh   = tid & 1u;
    unsigned lane = (tid >> 1) & 31u;
    unsigned ct   = (tid >> 6) & 255u;
    unsigned kc   = tid >> 14;
    unsigned n    = ct * 16u + (lane & 15u);
    unsigned g    = n >> 10;           // which gate weight matrix
    unsigned r    = n & 1023u;         // row within that matrix
    unsigned k0   = kc * 32u + ((lane >> 4) * 8u) + hh * 16u;

    const float* W = (g == 0) ? Wf : (g == 1) ? Wu : (g == 2) ? Wc : Wo;
    const float* src = W + (size_t)r * 2048u + k0;

    unsigned short t[8];
#pragma unroll
    for (int i = 0; i < 8; ++i) t[i] = f2bf(__builtin_nontemporal_load(src + i));

    uint4 v;
    v.x = (unsigned)t[0] | ((unsigned)t[1] << 16);
    v.y = (unsigned)t[2] | ((unsigned)t[3] << 16);
    v.z = (unsigned)t[4] | ((unsigned)t[5] << 16);
    v.w = (unsigned)t[6] | ((unsigned)t[7] << 16);
    size_t dst = (((size_t)kc * CT_TOT + ct) * 32u + lane) * 16u + hh * 8u;
    *(uint4*)(Bp + dst) = v;
}

// ---------------------------------------------------------------------------
// GEMM + fused LSTM epilogue.
// Grid: (128, 32). Block 256 = 8 waves.
// Block tile: rows blockIdx.x*128..+127 (row tiles rt0..rt0+7),
//             gate cols j = blockIdx.y*32..+31 in each of 4 gates
//             (col tiles ct = g*64 + jb*2 + {0,1} -> frags f=0..7).
// Wave w owns row tile rt0+w and all 8 col fragments: 8 WMMAs / K-step.
// Staging: GLOBAL_LOAD_ASYNC_TO_LDS_B128, double-buffered (2 x 16KB LDS),
// one s_barrier per K-step. Loop body is branch-free: chunk (kc+1)&63 is
// issued unconditionally (kc=63 re-stages chunk 0 into the dead buffer,
// which is never read; s_endpgm drains ASYNCcnt).
// ---------------------------------------------------------------------------
__global__ __launch_bounds__(256, 2) void lstm_gemm_kernel(
    const unsigned short* __restrict__ Ap, const unsigned short* __restrict__ Bp,
    const float* __restrict__ bf_, const float* __restrict__ bu_,
    const float* __restrict__ bc_, const float* __restrict__ bo_,
    const float* __restrict__ cin, float* __restrict__ out)
{
    // [buf][ A: 8KB | B: 8KB ] x2 = 32KB
    __shared__ __align__(32) unsigned short lds[16384];

    const unsigned tid  = threadIdx.x;
    const unsigned wave = tid >> 5;
    const unsigned lane = tid & 31u;
    const unsigned rt0  = blockIdx.x * 8u;
    const unsigned jb   = blockIdx.y;
    const unsigned seg  = tid >> 6;      // gate segment (2KB of B) this thread stages
    const unsigned so   = tid & 63u;

    const unsigned lds_base = (unsigned)(size_t)(&lds[0]);   // LDS byte offset

    v8f acc[8];
#pragma unroll
    for (int f = 0; f < 8; ++f)
#pragma unroll
        for (int i = 0; i < 8; ++i) acc[f][i] = 0.0f;

    // Issue async staging of K-chunk kc into buffer `buf` (4 x b128 per thread).
    auto issue_chunk = [&](unsigned kc, unsigned buf) {
        const char* gA = (const char*)Ap + ((size_t)kc * RT_TOT + rt0) * 1024u
                         + tid * 32u;
        const char* gB = (const char*)Bp +
                         ((size_t)kc * CT_TOT + seg * 64u + jb * 2u) * 1024u
                         + so * 32u;
        unsigned la = lds_base + buf * 16384u + tid * 32u;                      // A region
        unsigned lb = lds_base + buf * 16384u + 8192u + (seg * 64u + so) * 32u; // B region
        async_b128    (la, gA);
        async_b128_o16(la, gA);   // +16 on both LDS and global via INST_OFFSET
        async_b128    (lb, gB);
        async_b128_o16(lb, gB);
    };

    issue_chunk(0u, 0u);

    for (unsigned kc = 0; kc < KCHUNKS; ++kc) {
        const unsigned cur = kc & 1u;
        wait_async0();       // my chunk-kc fragments are in LDS
        __syncthreads();     // everyone's are; everyone done reading buf cur^1
        issue_chunk((kc + 1u) & (KCHUNKS - 1u), cur ^ 1u);  // overlaps WMMAs below

        const unsigned short* bufA = lds + (size_t)cur * 8192u;
        const unsigned short* bufB = bufA + 4096u;
        v16bf afrag = *(const v16bf*)(bufA + (wave * 32u + lane) * 16u);
#pragma unroll
        for (int f = 0; f < 8; ++f) {
            v16bf bfrag = *(const v16bf*)(bufB + ((unsigned)f * 32u + lane) * 16u);
            acc[f] = __builtin_amdgcn_wmma_f32_16x16x32_bf16(
                false, afrag, false, bfrag, (short)0, acc[f], false, false);
        }
    }

    // ---- fused LSTM epilogue ----
    // C layout: element (VGPR r, lane l) = C[M = r + 8*(l>>4)][N = l&15].
    // f=acc[s], u=acc[2+s], c~=acc[4+s], o=acc[6+s]; col = jb*32 + 16*s + (l&15).
    const unsigned jcol = jb * 32u + (lane & 15u);
    float Bf0 = bf_[jcol], Bf1 = bf_[jcol + 16u];
    float Bu0 = bu_[jcol], Bu1 = bu_[jcol + 16u];
    float Bc0 = bc_[jcol], Bc1 = bc_[jcol + 16u];
    float Bo0 = bo_[jcol], Bo1 = bo_[jcol + 16u];

    const unsigned row0 = blockIdx.x * 128u + wave * 16u + ((lane >> 4) * 8u);
    const size_t   hc_off = (size_t)BATCH * OUTF;   // c_new follows h_new in d_out

#pragma unroll
    for (int r = 0; r < 8; ++r) {
        const size_t base = (size_t)(row0 + (unsigned)r) * OUTF;
#pragma unroll
        for (int s = 0; s < 2; ++s) {
            unsigned col = jcol + 16u * (unsigned)s;
            float gf = acc[0 + s][r] + (s ? Bf1 : Bf0);
            float gu = acc[2 + s][r] + (s ? Bu1 : Bu0);
            float gc = acc[4 + s][r] + (s ? Bc1 : Bc0);
            float go = acc[6 + s][r] + (s ? Bo1 : Bo0);
            float F  = sigf(gf);
            float U  = sigf(gu);
            float Ct = tanh_fast(gc);
            float O  = sigf(go);
            float cn = F * __builtin_nontemporal_load(cin + base + col) + U * Ct;
            float hn = tanh_fast(cn) * O;
            __builtin_nontemporal_store(hn, out + base + col);            // h_new
            __builtin_nontemporal_store(cn, out + hc_off + base + col);   // c_new
        }
    }
}

// ---------------------------------------------------------------------------
extern "C" void kernel_launch(void* const* d_in, const int* in_sizes, int n_in,
                              void* d_out, int out_size, void* d_ws, size_t ws_size,
                              hipStream_t stream)
{
    const float* x   = (const float*)d_in[0];
    const float* h   = (const float*)d_in[1];
    const float* c   = (const float*)d_in[2];
    const float* Wf  = (const float*)d_in[3];
    const float* bf_ = (const float*)d_in[4];
    const float* Wu  = (const float*)d_in[5];
    const float* bu_ = (const float*)d_in[6];
    const float* Wc  = (const float*)d_in[7];
    const float* bc_ = (const float*)d_in[8];
    const float* Wo  = (const float*)d_in[9];
    const float* bo_ = (const float*)d_in[10];

    unsigned short* Ap = (unsigned short*)d_ws;                                 // 64 MB
    unsigned short* Bp = (unsigned short*)((char*)d_ws + (((size_t)64) << 20)); // +64MB
    (void)in_sizes; (void)n_in; (void)out_size; (void)ws_size;

    pack_A_kernel<<<16384, 256, 0, stream>>>(x, h, Ap);
    pack_B_kernel<<<4096, 256, 0, stream>>>(Wf, Wu, Wc, Wo, Bp);

    dim3 grid(BATCH / 128, OUTF / 32);   // (128, 32)
    lstm_gemm_kernel<<<grid, 256, 0, stream>>>(Ap, Bp, bf_, bu_, bc_, bo_, c, (float*)d_out);
}
